// Encoder_14121852469955
// MI455X (gfx1250) — matrix operand order repaired and checked by reading, not compiled, not run on hardware
//
#include <hip/hip_runtime.h>

// ---------------------------------------------------------------------------
// Transformer encoder (B=2,S=2048,D=1024,H=16,HD=64,F=4096,L=6) for gfx1250.
// All GEMM-shaped math runs on v_wmma_f32_16x16x32_bf16 (bf16 in, fp32 acc).
// Round 2: double-buffered GEMM staging, packed b64 LDS stores, L2 prefetch.
// ---------------------------------------------------------------------------

typedef __attribute__((ext_vector_type(16))) __bf16 v16bf;
typedef __attribute__((ext_vector_type(8)))  float  v8f;

#define CB  2
#define CS  2048
#define CD  1024
#define CH  16
#define CHD 64
#define CL  6
#define CF  4096
#define CM  (CB*CS)       // 4096 rows of activations

__device__ __forceinline__ unsigned int f2bfbits(float f) {
    unsigned int u = __float_as_uint(f);
    u += 0x7FFFu + ((u >> 16) & 1u);      // round-to-nearest-even
    return u >> 16;
}
__device__ __forceinline__ unsigned short f2bf(float f) {
    return (unsigned short)f2bfbits(f);
}
// pack float4 -> 4 bf16 (one uint2 / ds_store_b64)
__device__ __forceinline__ uint2 pk4(float4 f) {
    uint2 r;
    r.x = f2bfbits(f.x) | (f2bfbits(f.y) << 16);
    r.y = f2bfbits(f.z) | (f2bfbits(f.w) << 16);
    return r;
}

__device__ __forceinline__ v16bf frag_from(const unsigned short* lo,
                                           const unsigned short* hi) {
    union { uint4 u[2]; v16bf v; } t;
    t.u[0] = *(const uint4*)lo;
    t.u[1] = *(const uint4*)hi;
    return t.v;
}

__device__ __forceinline__ float expg(float x) {
    return (x < -80.0f) ? 0.0f : __expf(x);
}

// ---------------------------------------------------------------------------
// Embedding + sinusoidal positional encoding.  grid = B*S rows, block = 256.
// ---------------------------------------------------------------------------
__global__ __launch_bounds__(256)
void embed_kernel(const int* __restrict__ tok, const float* __restrict__ emb,
                  float* __restrict__ x) {
    const int row = blockIdx.x;            // b*S + s
    const int s   = row % CS;
    const int t   = tok[row];
    for (int d = threadIdx.x; d < CD; d += 256) {
        const int   i   = d >> 1;
        const float div = __expf((float)(2 * i) * (-9.210340371976184f / (float)CD));
        const float arg = (float)s * div;
        const float pe  = (d & 1) ? __cosf(arg) : __sinf(arg);
        x[(size_t)row * CD + d] = emb[(size_t)t * CD + d] + pe;
    }
}

// ---------------------------------------------------------------------------
// Y[M,N] = act(X[M,K] @ W[K,N] + bias[N]).  fp32 in/out, bf16 WMMA inside.
// Block tile 128x128, K step 32. 8 waves in 2x4 grid; wave tile 64x32.
// Double-buffered: next K-tile is fetched to registers under the WMMAs.
// ---------------------------------------------------------------------------
template <int RELU>
__global__ __launch_bounds__(256)
void gemm_bias(const float* __restrict__ A, const float* __restrict__ W,
               const float* __restrict__ bias, float* __restrict__ Y,
               int K, int N) {
    __shared__ __align__(16) unsigned short As[128 * 40];  // [m][k], stride 40
    __shared__ __align__(16) unsigned short Bs[128 * 40];  // [n][k] (W^T), stride 40

    const int tid  = threadIdx.x;
    const int lane = tid & 31, wave = tid >> 5;
    const int half = lane >> 4, l16 = lane & 15;
    const int wRow = wave >> 2, wCol = wave & 3;           // 2 x 4 wave grid
    const int bM = blockIdx.y * 128, bN = blockIdx.x * 128;

    // per-thread staging coordinates (4 float4 each for A and W tiles)
    int ar[4], ac[4], wk[4], wn[4];
    for (int it = 0; it < 4; ++it) {
        const int idx = (it * 256 + tid) * 4;
        ar[it] = idx >> 5;  ac[it] = idx & 31;    // A tile 128x32
        wk[it] = idx >> 7;  wn[it] = idx & 127;   // W tile 32x128
    }

    v8f acc[4][2] = {};
    float4 pa[4], pb[4];

    // prologue: fetch K-tile 0
    for (int it = 0; it < 4; ++it) {
        pa[it] = *(const float4*)(A + (size_t)(bM + ar[it]) * K + ac[it]);
        pb[it] = *(const float4*)(W + (size_t)wk[it] * N + bN + wn[it]);
    }

    for (int k0 = 0; k0 < K; k0 += 32) {
        // stage current tile (registers -> LDS as bf16)
        for (int it = 0; it < 4; ++it)
            *(uint2*)&As[ar[it] * 40 + ac[it]] = pk4(pa[it]);
        for (int it = 0; it < 4; ++it) {
            const int nc = wn[it], kk = wk[it];
            Bs[(nc + 0) * 40 + kk] = f2bf(pb[it].x);
            Bs[(nc + 1) * 40 + kk] = f2bf(pb[it].y);
            Bs[(nc + 2) * 40 + kk] = f2bf(pb[it].z);
            Bs[(nc + 3) * 40 + kk] = f2bf(pb[it].w);
        }
        __syncthreads();

        // fetch next K-tile while WMMAs run on the current one
        const int kn = k0 + 32;
        if (kn < K) {
            for (int it = 0; it < 4; ++it) {
                pa[it] = *(const float4*)(A + (size_t)(bM + ar[it]) * K + kn + ac[it]);
                pb[it] = *(const float4*)(W + (size_t)(kn + wk[it]) * N + bN + wn[it]);
            }
            if (kn + 32 < K)   // keep the weight stream warm in L2
                __builtin_prefetch(W + (size_t)(kn + 32 + wk[0]) * N + bN + wn[0], 0, 1);
        }

        v16bf bfr[2];
        for (int j = 0; j < 2; ++j) {
            const int col = wCol * 32 + j * 16 + l16;
            const unsigned short* p = &Bs[col * 40 + half * 16];
            bfr[j] = frag_from(p, p + 8);
        }
        for (int i = 0; i < 4; ++i) {
            const int row = wRow * 64 + i * 16 + l16;
            const unsigned short* p = &As[row * 40 + half * 16];
            const v16bf af = frag_from(p, p + 8);
            for (int j = 0; j < 2; ++j)
                acc[i][j] = __builtin_amdgcn_wmma_f32_16x16x32_bf16(
                    false, af, false, bfr[j], (short)0, acc[i][j], false, false);
        }
        __syncthreads();
    }

    for (int j = 0; j < 2; ++j) {
        const int col = bN + wCol * 32 + j * 16 + l16;
        const float bv = bias[col];
        for (int i = 0; i < 4; ++i) {
            const int rbase = bM + wRow * 64 + i * 16 + half * 8;
            for (int vi = 0; vi < 8; ++vi) {
                float o = acc[i][j][vi] + bv;
                if (RELU) o = fmaxf(o, 0.0f);
                Y[(size_t)(rbase + vi) * N + col] = o;
            }
        }
    }
}

// ---------------------------------------------------------------------------
// Flash attention. q/k/v are [B*S, D] with head h at cols h*64..h*64+63.
// grid = (S/128, B*H); block = 256 (8 waves, 16 Q rows per wave).
// Online softmax, bf16 WMMA for Q@K^T and P@V, fp32 accumulation.
// ---------------------------------------------------------------------------
__global__ __launch_bounds__(256)
void flash_attn(const float* __restrict__ Q, const float* __restrict__ Km,
                const float* __restrict__ V, const int* __restrict__ mask,
                float* __restrict__ O) {
    __shared__ __align__(16) unsigned short Qs[128 * 72];       // [qrow][hd]
    __shared__ __align__(16) unsigned short Ks[64 * 72];        // [key][hd]
    __shared__ __align__(16) unsigned short Vt[64 * 72];        // [hd][key]
    __shared__ __align__(16) unsigned short Ps[8 * 16 * 72];    // per-wave P

    const int tid  = threadIdx.x, lane = tid & 31, wave = tid >> 5;
    const int half = lane >> 4, l16 = lane & 15;
    const int bh = blockIdx.y, b = bh / CH, h = bh % CH;
    const int q0 = blockIdx.x * 128;
    const float scale = 0.125f;                     // 1/sqrt(HD)
    const int* mrow = mask + (size_t)b * CS;

    // stage Q tile (128 x 64) as bf16
    for (int it = 0; it < 8; ++it) {
        const int idx = (it * 256 + tid) * 4;
        const int r = idx >> 6, c = idx & 63;
        const float4 f = *(const float4*)(Q + (size_t)(b * CS + q0 + r) * CD + h * 64 + c);
        *(uint2*)&Qs[r * 72 + c] = pk4(f);
    }
    __syncthreads();

    v16bf qf[2];
    for (int st = 0; st < 2; ++st) {
        const unsigned short* p = &Qs[(wave * 16 + l16) * 72 + st * 32 + half * 16];
        qf[st] = frag_from(p, p + 8);
    }

    v8f   cacc[4] = {};
    float mr[8], lr[8];
    for (int i = 0; i < 8; ++i) { mr[i] = -3.0e38f; lr[i] = 0.0f; }

    for (int kb = 0; kb < CS; kb += 64) {
        __syncthreads();
        // stage K (row-major, packed b64) and V (transposed) chunks
        for (int it = 0; it < 4; ++it) {
            const int idx = (it * 256 + tid) * 4;
            const int r = idx >> 6, c = idx & 63;
            const size_t gr = (size_t)(b * CS + kb + r) * CD + h * 64 + c;
            const float4 fk = *(const float4*)(Km + gr);
            *(uint2*)&Ks[r * 72 + c] = pk4(fk);
            const float4 fv = *(const float4*)(V + gr);
            Vt[(c + 0) * 72 + r] = f2bf(fv.x);
            Vt[(c + 1) * 72 + r] = f2bf(fv.y);
            Vt[(c + 2) * 72 + r] = f2bf(fv.z);
            Vt[(c + 3) * 72 + r] = f2bf(fv.w);
        }
        if (kb + 64 < CS)
            __builtin_prefetch(Km + (size_t)(b * CS + kb + 64) * CD + h * 64, 0, 1);
        __syncthreads();

        // scores: 4 tiles of 16 keys, K-dim = 64 (2 WMMA steps each)
        float sc[4][8];
        for (int t = 0; t < 4; ++t) {
            v8f sacc = {};
            const unsigned short* p0 = &Ks[(t * 16 + l16) * 72 + half * 16];
            const unsigned short* p1 = &Ks[(t * 16 + l16) * 72 + 32 + half * 16];
            const v16bf kf0 = frag_from(p0, p0 + 8);
            const v16bf kf1 = frag_from(p1, p1 + 8);
            sacc = __builtin_amdgcn_wmma_f32_16x16x32_bf16(false, qf[0], false, kf0,
                                                           (short)0, sacc, false, false);
            sacc = __builtin_amdgcn_wmma_f32_16x16x32_bf16(false, qf[1], false, kf1,
                                                           (short)0, sacc, false, false);
            const int key = kb + t * 16 + l16;
            const bool masked = (mrow[key] == 0);
            for (int vi = 0; vi < 8; ++vi)
                sc[t][vi] = masked ? -1e9f : sacc[vi] * scale;
        }

        // online softmax (row = vi + 8*half; reductions stay inside 16-lane half)
        for (int vi = 0; vi < 8; ++vi) {
            float pm = fmaxf(fmaxf(sc[0][vi], sc[1][vi]), fmaxf(sc[2][vi], sc[3][vi]));
            for (int off = 8; off >= 1; off >>= 1)
                pm = fmaxf(pm, __shfl_xor(pm, off, 32));
            const float mn = fmaxf(mr[vi], pm);
            const float fct = expg(mr[vi] - mn);
            float rs = 0.0f;
            for (int t = 0; t < 4; ++t) {
                const float p = expg(sc[t][vi] - mn);
                sc[t][vi] = p;
                rs += p;
            }
            for (int off = 8; off >= 1; off >>= 1)
                rs += __shfl_xor(rs, off, 32);
            lr[vi] = lr[vi] * fct + rs;
            mr[vi] = mn;
            for (int j = 0; j < 4; ++j) cacc[j][vi] *= fct;
        }

        // spill P (C layout) to per-wave LDS as bf16, reload in A layout
        for (int vi = 0; vi < 8; ++vi) {
            const int m = vi + half * 8;
            for (int t = 0; t < 4; ++t)
                Ps[wave * 16 * 72 + m * 72 + t * 16 + l16] = f2bf(sc[t][vi]);
        }
        __syncthreads();

        for (int st = 0; st < 2; ++st) {
            const unsigned short* pp = &Ps[wave * 16 * 72 + l16 * 72 + st * 32 + half * 16];
            const v16bf pf = frag_from(pp, pp + 8);
            for (int j = 0; j < 4; ++j) {
                const unsigned short* vp = &Vt[(j * 16 + l16) * 72 + st * 32 + half * 16];
                const v16bf vf = frag_from(vp, vp + 8);
                cacc[j] = __builtin_amdgcn_wmma_f32_16x16x32_bf16(
                    false, pf, false, vf, (short)0, cacc[j], false, false);
            }
        }
    }

    // normalize and write ctx in head-concatenated layout [B*S, D]
    for (int j = 0; j < 4; ++j) {
        const int col = h * 64 + j * 16 + l16;
        for (int vi = 0; vi < 8; ++vi) {
            const int row = q0 + wave * 16 + vi + half * 8;
            O[(size_t)(b * CS + row) * CD + col] = cacc[j][vi] / lr[vi];
        }
    }
}

// ---------------------------------------------------------------------------
// out = LayerNorm(resid + y) * g + be.  One block (256 thr) per row, D=1024.
// ---------------------------------------------------------------------------
__global__ __launch_bounds__(256)
void add_ln(const float* __restrict__ R, const float* __restrict__ Yv,
            const float* __restrict__ g, const float* __restrict__ be,
            float* __restrict__ Out) {
    __shared__ float red1[256];
    __shared__ float red2[256];
    const int row = blockIdx.x, tid = threadIdx.x;
    float x[4], s1 = 0.0f, s2 = 0.0f;
    for (int i = 0; i < 4; ++i) {
        const int c = tid + i * 256;
        const float v = R[(size_t)row * CD + c] + Yv[(size_t)row * CD + c];
        x[i] = v; s1 += v; s2 += v * v;
    }
    red1[tid] = s1; red2[tid] = s2;
    __syncthreads();
    for (int off = 128; off >= 1; off >>= 1) {
        if (tid < off) { red1[tid] += red1[tid + off]; red2[tid] += red2[tid + off]; }
        __syncthreads();
    }
    const float mean = red1[0] * (1.0f / CD);
    const float var  = red2[0] * (1.0f / CD) - mean * mean;
    const float inv  = rsqrtf(var + 1e-5f);
    for (int i = 0; i < 4; ++i) {
        const int c = tid + i * 256;
        Out[(size_t)row * CD + c] = (x[i] - mean) * inv * g[c] + be[c];
    }
}

// ---------------------------------------------------------------------------
extern "C" void kernel_launch(void* const* d_in, const int* in_sizes, int n_in,
                              void* d_out, int out_size, void* d_ws, size_t ws_size,
                              hipStream_t stream) {
    const int*   tokens = (const int*)d_in[0];
    const int*   mask   = (const int*)d_in[1];
    const float* emb    = (const float*)d_in[2];
    const float* Wq = (const float*)d_in[3];  const float* bq = (const float*)d_in[4];
    const float* Wk = (const float*)d_in[5];  const float* bk = (const float*)d_in[6];
    const float* Wv = (const float*)d_in[7];  const float* bv = (const float*)d_in[8];
    const float* Wo = (const float*)d_in[9];  const float* bo = (const float*)d_in[10];
    const float* W1 = (const float*)d_in[11]; const float* b1 = (const float*)d_in[12];
    const float* W2 = (const float*)d_in[13]; const float* b2 = (const float*)d_in[14];
    const float* g1 = (const float*)d_in[15]; const float* be1 = (const float*)d_in[16];
    const float* g2 = (const float*)d_in[17]; const float* be2 = (const float*)d_in[18];

    float* x = (float*)d_out;                 // running activations [M, D]

    // workspace layout (floats): q,k,v,ctx,tmp = M*D each; h1 = M*F  (~144 MB)
    float* q   = (float*)d_ws;
    float* k   = q   + (size_t)CM * CD;
    float* v   = k   + (size_t)CM * CD;
    float* ctx = v   + (size_t)CM * CD;
    float* tmp = ctx + (size_t)CM * CD;
    float* h1  = tmp + (size_t)CM * CD;

    embed_kernel<<<CM, 256, 0, stream>>>(tokens, emb, x);

    const dim3 gProjD(CD / 128, CM / 128);    // N=1024 GEMMs
    const dim3 gFF1(CF / 128, CM / 128);      // N=4096 GEMM
    const dim3 gAttn(CS / 128, CB * CH);

    for (int l = 0; l < CL; ++l) {
        const size_t oW = (size_t)l * CD * CD, oB = (size_t)l * CD;
        gemm_bias<0><<<gProjD, 256, 0, stream>>>(x, Wq + oW, bq + oB, q, CD, CD);
        gemm_bias<0><<<gProjD, 256, 0, stream>>>(x, Wk + oW, bk + oB, k, CD, CD);
        gemm_bias<0><<<gProjD, 256, 0, stream>>>(x, Wv + oW, bv + oB, v, CD, CD);

        flash_attn<<<gAttn, 256, 0, stream>>>(q, k, v, mask, ctx);

        gemm_bias<0><<<gProjD, 256, 0, stream>>>(ctx, Wo + oW, bo + oB, tmp, CD, CD);
        add_ln<<<CM, 256, 0, stream>>>(x, tmp, g1 + oB, be1 + oB, x);

        gemm_bias<1><<<gFF1, 256, 0, stream>>>(x, W1 + (size_t)l * CD * CF,
                                               b1 + (size_t)l * CF, h1, CD, CF);
        gemm_bias<0><<<gProjD, 256, 0, stream>>>(h1, W2 + (size_t)l * CF * CD,
                                                 b2 + oB, tmp, CF, CD);
        add_ln<<<CM, 256, 0, stream>>>(x, tmp, g2 + oB, be2 + oB, x);
    }
}
